// LIFNet_29798483100215
// MI455X (gfx1250) — matrix operand order repaired and checked by reading, not compiled, not run on hardware
//
#include <hip/hip_runtime.h>

// Fused LIF SNN for MI455X (gfx1250, wave32):
//   layer1 GEMM  : V_WMMA_F32_16X16X4_F32 (full fp32, K=784), K-slabs double-
//                  buffered in LDS via GLOBAL_LOAD_ASYNC_TO_LDS_B128/B64
//                  (ASYNCcnt-tracked, overlapped with WMMA compute).
//   layer2 GEMM  : V_WMMA_F32_16X16X32_BF16 with hi/lo split W2 (~fp32
//                  accuracy; A = spikes, exact in bf16).
//   25 timesteps of LIF recurrence fused on-chip; mem1 lives in the WMMA
//   C-fragment registers, spikes pass through LDS each step.

typedef float  v2f   __attribute__((ext_vector_type(2)));
typedef float  v8f   __attribute__((ext_vector_type(8)));
typedef __bf16 v16bf __attribute__((ext_vector_type(16)));
typedef int    v4i_  __attribute__((vector_size(16)));   // async b128 payload
typedef int    v2i_  __attribute__((vector_size(8)));    // async b64 payload

#define D_IN   784
#define H_DIM  256
#define O_DIM  10
#define TSTEPS 25
#define BETA   0.95f
#define B_ROWS 32            // batch rows per block
#define KC     16            // K-slab (784 = 49*16)
#define KCP    20            // padded slab row stride (16B aligned, conflict-free)
#define NSLAB  (D_IN / KC)   // 49
#define SPKP   264           // spike buffer row stride (u16), conflict-free

#if __has_builtin(__builtin_amdgcn_global_load_async_to_lds_b128) && \
    __has_builtin(__builtin_amdgcn_global_load_async_to_lds_b64)
#define HAVE_ASYNC 1
#else
#define HAVE_ASYNC 0
#endif

#define AS1 __attribute__((address_space(1)))
#define AS3 __attribute__((address_space(3)))

union BF16Frag { unsigned u[8]; v16bf v; };

__device__ __forceinline__ unsigned short f2bf(float f) {
  unsigned u = __float_as_uint(f);
  u += 0x7FFFu + ((u >> 16) & 1u);        // round-to-nearest-even
  return (unsigned short)(u >> 16);
}

#if HAVE_ASYNC
__device__ __forceinline__ void cp_async_b128(void* lds, const void* g) {
  // global AS1 numeric address == generic numeric address for global memory
  __builtin_amdgcn_global_load_async_to_lds_b128(
      (AS1 v4i_*)(uintptr_t)g, (AS3 v4i_*)lds, 0, 0);
}
__device__ __forceinline__ void cp_async_b64(void* lds, const void* g) {
  __builtin_amdgcn_global_load_async_to_lds_b64(
      (AS1 v2i_*)(uintptr_t)g, (AS3 v2i_*)lds, 0, 0);
}
#endif

__device__ __forceinline__ void wait_async_le5() {
#if HAVE_ASYNC
#if __has_builtin(__builtin_amdgcn_s_wait_asynccnt)
  __builtin_amdgcn_s_wait_asynccnt(5);
#else
  asm volatile("s_wait_asynccnt 0x5" ::: "memory");
#endif
#endif
}
__device__ __forceinline__ void wait_async_le0() {
#if HAVE_ASYNC
#if __has_builtin(__builtin_amdgcn_s_wait_asynccnt)
  __builtin_amdgcn_s_wait_asynccnt(0);
#else
  asm volatile("s_wait_asynccnt 0x0" ::: "memory");
#endif
#endif
}

__global__ __launch_bounds__(256) void lif_fused(
    const float* __restrict__ x,  const float* __restrict__ W1,
    const float* __restrict__ b1, const float* __restrict__ W2,
    const float* __restrict__ b2, float* __restrict__ out)
{
  __shared__ float          wt[2][H_DIM * KCP];      // 2 x 20 KB  W1 slab [n][k]
  __shared__ float          xs[2][B_ROWS * KCP];     // 2 x 2.5 KB x slab  [m][k]
  __shared__ unsigned short spk[B_ROWS * SPKP];      // 16.5 KB bf16 spikes
  __shared__ unsigned       w2h[(H_DIM / 2) * 16];   // 8 KB packed bf16 k-pairs
  __shared__ unsigned       w2l[(H_DIM / 2) * 16];   // 8 KB (lo residual)

  const int t    = threadIdx.x;
  const int wave = t >> 5;
  const int lane = t & 31;
  const int half = (lane >= 16) ? 1 : 0;
  const int l15  = lane & 15;
  const int mt   = wave & 1;             // m-subtile (16 rows)
  const int nt0  = (wave >> 1) * 4;      // first of 4 n-tiles
  const int row0 = blockIdx.x * B_ROWS;

  // stage one K-slab (5 uniform async ops per thread, or sync fallback)
  auto stage = [&](int slab) {
    const int kc  = slab * KC;
    float* wb = &wt[slab & 1][0];
    float* xb = &xs[slab & 1][0];
    const float* wsrc = W1 + (size_t)t * D_IN + kc;
    const int r = t >> 3, ko = (t & 7) * 2;
    const float* xsrc = x + (size_t)(row0 + r) * D_IN + kc + ko;
#if HAVE_ASYNC
    #pragma unroll
    for (int j = 0; j < 4; ++j)
      cp_async_b128(&wb[t * KCP + 4 * j], wsrc + 4 * j);
    cp_async_b64(&xb[r * KCP + ko], xsrc);
#else
    #pragma unroll
    for (int j = 0; j < 4; ++j)
      *(float4*)&wb[t * KCP + 4 * j] = *(const float4*)(wsrc + 4 * j);
    xb[r * KCP + ko]     = xsrc[0];
    xb[r * KCP + ko + 1] = xsrc[1];
#endif
  };

  // ---- stage W2 hi/lo bf16 split into LDS, packed as [kpair][n] dwords ----
  {
    int base = t * 8;                    // 2048 entries / 256 threads
    #pragma unroll
    for (int i = 0; i < 8; ++i) {
      int idx = base + i;
      int kp = idx >> 4, n = idx & 15;
      float f0 = (n < O_DIM) ? W2[n * H_DIM + 2 * kp]     : 0.f;
      float f1 = (n < O_DIM) ? W2[n * H_DIM + 2 * kp + 1] : 0.f;
      unsigned short h0 = f2bf(f0), h1 = f2bf(f1);
      float r0 = f0 - __uint_as_float((unsigned)h0 << 16);
      float r1 = f1 - __uint_as_float((unsigned)h1 << 16);
      w2h[idx] = (unsigned)h0 | ((unsigned)h1 << 16);
      w2l[idx] = (unsigned)f2bf(r0) | ((unsigned)f2bf(r1) << 16);
    }
  }

  // ---- layer-1 GEMM: cur1[32,256] = x @ W1^T via f32 WMMA, double-buffered ----
  v8f c[4];
  #pragma unroll
  for (int j = 0; j < 4; ++j)
    #pragma unroll
    for (int v = 0; v < 8; ++v) c[j][v] = 0.f;

  stage(0);
  const int rowA = mt * 16 + l15;
  for (int i = 0; i < NSLAB; ++i) {
    if (i + 1 < NSLAB) { stage(i + 1); wait_async_le5(); }
    else               { wait_async_le0(); }
    __syncthreads();
    const float* xb = &xs[i & 1][0];
    const float* wb = &wt[i & 1][0];
    #pragma unroll
    for (int ks = 0; ks < KC / 4; ++ks) {
      int kk = ks * 4 + half * 2;
      v2f a = *(const v2f*)&xb[rowA * KCP + kk];         // A: K=kk,kk+1
      #pragma unroll
      for (int j = 0; j < 4; ++j) {
        int n = (nt0 + j) * 16 + l15;
        v2f b = *(const v2f*)&wb[n * KCP + kk];          // single ds_load_b64
        c[j] = __builtin_amdgcn_wmma_f32_16x16x4_f32(
                   false, a, false, b, (short)0, c[j], false, false);
      }
    }
    __syncthreads();   // slab consumed; safe to overwrite this buffer next+1
  }

  // add b1 -> cur1 (kept in registers; same fragment layout reused for mem1)
  v8f cur1[4], mem1[4];
  #pragma unroll
  for (int j = 0; j < 4; ++j) {
    float bv = b1[(nt0 + j) * 16 + l15];
    cur1[j] = c[j] + bv;
    #pragma unroll
    for (int v = 0; v < 8; ++v) mem1[j][v] = 0.f;
  }

  // layer-2 state (waves 0/1 only)
  v8f m2, cnt;
  #pragma unroll
  for (int v = 0; v < 8; ++v) { m2[v] = 0.f; cnt[v] = 0.f; }
  float b2v = (l15 < O_DIM) ? b2[l15] : 0.f;

  // ---- 25-step LIF recurrence ----
  for (int step = 0; step < TSTEPS; ++step) {
    // LIF layer 1 (reset from PRE-update mem, subtract mechanism) + spike write
    #pragma unroll
    for (int j = 0; j < 4; ++j) {
      int ccol = (nt0 + j) * 16 + l15;
      #pragma unroll
      for (int v = 0; v < 8; ++v) {
        float m = mem1[j][v];
        float r = (m > 1.f) ? 1.f : 0.f;
        m = BETA * m + cur1[j][v] - r;
        mem1[j][v] = m;
        int rr = mt * 16 + v + half * 8;                 // C-layout row
        spk[rr * SPKP + ccol] = (m > 1.f) ? (unsigned short)0x3F80
                                          : (unsigned short)0;
      }
    }
    __syncthreads();

    if (wave < 2) {  // wave-uniform branch: EXEC all-ones inside (WMMA legal)
      v8f c2;
      #pragma unroll
      for (int v = 0; v < 8; ++v) c2[v] = b2v;
      const int rowS = wave * 16 + l15;
      #pragma unroll
      for (int ks = 0; ks < 8; ++ks) {
        BF16Frag A, Bh, Bl;
        #pragma unroll
        for (int v = 0; v < 8; ++v) {
          int kp = (v & 3) * 2 + (v >> 2) * 16 + half * 8;   // 16-bit A layout
          A.u[v]  = *(const unsigned*)&spk[rowS * SPKP + ks * 32 + kp];
          int kb  = ks * 16 + (v & 3) + (v >> 2) * 8 + half * 4;  // 16-bit B layout
          Bh.u[v] = w2h[kb * 16 + l15];
          Bl.u[v] = w2l[kb * 16 + l15];
        }
        c2 = __builtin_amdgcn_wmma_f32_16x16x32_bf16(
                 false, A.v, false, Bh.v, (short)0, c2, false, false);
        c2 = __builtin_amdgcn_wmma_f32_16x16x32_bf16(
                 false, A.v, false, Bl.v, (short)0, c2, false, false);
      }
      #pragma unroll
      for (int v = 0; v < 8; ++v) {
        float m = m2[v];
        float r = (m > 1.f) ? 1.f : 0.f;
        m = BETA * m + c2[v] - r;
        m2[v] = m;
        cnt[v] += (m > 1.f) ? 1.f : 0.f;
      }
    }
    __syncthreads();   // spikes fully consumed before next step overwrites
  }

  // ---- write spike counts [B,10] ----
  if (wave < 2 && l15 < O_DIM) {
    #pragma unroll
    for (int v = 0; v < 8; ++v) {
      int rr = wave * 16 + v + half * 8;
      out[(size_t)(row0 + rr) * O_DIM + l15] = cnt[v];
    }
  }
}

extern "C" void kernel_launch(void* const* d_in, const int* in_sizes, int n_in,
                              void* d_out, int out_size, void* d_ws, size_t ws_size,
                              hipStream_t stream) {
  const float* x  = (const float*)d_in[0];
  const float* W1 = (const float*)d_in[1];
  const float* b1 = (const float*)d_in[2];
  const float* W2 = (const float*)d_in[3];
  const float* b2 = (const float*)d_in[4];
  float* out = (float*)d_out;
  const int B = in_sizes[0] / D_IN;              // 16384
  lif_fused<<<B / B_ROWS, 256, 0, stream>>>(x, W1, b1, W2, b2, out);
}
